// MiniAttentionLayer_40673340293687
// MI455X (gfx1250) — matrix operand order repaired and checked by reading, not compiled, not run on hardware
//
#include <hip/hip_runtime.h>
#include <stdint.h>

#define B_ROWS 32768
#define BM 64

typedef __attribute__((ext_vector_type(16))) __bf16 v16bf;
typedef __attribute__((ext_vector_type(8)))  float  v8f;

// ---------- bf16 helpers ----------
static __device__ __forceinline__ unsigned short f2us(float f){
  union { float f; unsigned int i; } u; u.f = f;
  unsigned int r = u.i + 0x7FFFu + ((u.i >> 16) & 1u);
  return (unsigned short)(r >> 16);
}
static __device__ __forceinline__ float us2f(unsigned int s){
  union { float f; unsigned int i; } u; u.i = (s & 0xFFFFu) << 16; return u.f;
}
static __device__ __forceinline__ unsigned int pk2(float a, float b){
  return (unsigned int)f2us(a) | ((unsigned int)f2us(b) << 16);
}

// ---------- WMMA fragment loaders (CDNA5 16x16x32 bf16 layouts) ----------
// A: 16x32 (MxK). lanes 0-15: M=lane, K in {k0..k0+7, k0+16..k0+23};
//    lanes 16-31: M=lane-16, K in {k0+8..k0+15, k0+24..k0+31}.
static __device__ __forceinline__ v16bf loadA(const unsigned short* p0, int lda,
                                              int m0, int k0, int lane){
  const unsigned short* p = p0 + (size_t)(m0 + (lane & 15)) * lda + k0 + ((lane >> 4) << 3);
  union { v16bf v; uint4 q[2]; } u;
  u.q[0] = *(const uint4*)p;
  u.q[1] = *(const uint4*)(p + 16);
  return u.v;
}
// B: 32x16 (KxN) built from row-major W[N][K]: lane<16 -> N=lane, K in {k0..k0+7, k0+16..k0+23};
//    lane>=16 -> N=lane-16, K in {k0+8..k0+15, k0+24..k0+31}.
static __device__ __forceinline__ v16bf loadB(const unsigned short* w, int ldk,
                                              int n0, int k0, int lane){
  const unsigned short* p = w + (size_t)(n0 + (lane & 15)) * ldk + k0 + ((lane >> 4) << 3);
  union { v16bf v; uint4 q[2]; } u;
  u.q[0] = *(const uint4*)p;
  u.q[1] = *(const uint4*)(p + 16);
  return u.v;
}
static __device__ __forceinline__ v8f wmma_bf16(v16bf a, v16bf b, v8f c){
  return __builtin_amdgcn_wmma_f32_16x16x32_bf16(false, a, false, b, (short)0, c, false, false);
}

// ---------- load 64xC fp32 tile from global, convert to bf16 in LDS ----------
static __device__ __forceinline__ void load_tile(const float* g, int row0, int cols,
                                                 unsigned short* sDst, int tid){
  int n4 = (BM * cols) >> 2;
  const float4* src = (const float4*)(g + (size_t)row0 * cols);
  uint2* dst = (uint2*)sDst;
  for (int i = tid; i < n4; i += 256){
    float4 f = src[i];
    uint2 o; o.x = pk2(f.x, f.y); o.y = pk2(f.z, f.w);
    dst[i] = o;
  }
}

// ---------- one fused GEMM stage: sD[64x512] = sA[64xK] @ W[512xK]^T + bias ----------
template<int K>
static __device__ __forceinline__ void gemm_tile_to_lds(const unsigned short* sA,
                                                        const unsigned short* W,
                                                        const float* bias,
                                                        unsigned short* sD,
                                                        int wave, int lane){
  constexpr int NK = K >> 5;
  int mb = (wave & 3) << 4;   // 4 m-blocks
  int nh = wave >> 2;         // 2 n-halves (16 n-blocks each)
  v16bf af[NK];
#pragma unroll
  for (int ks = 0; ks < NK; ++ks) af[ks] = loadA(sA, K, mb, ks << 5, lane);
#pragma unroll 4
  for (int nb = nh << 4; nb < (nh << 4) + 16; ++nb){
    int n0 = nb << 4;
    v8f c = {};
#pragma unroll
    for (int ks = 0; ks < NK; ++ks)
      c = wmma_bf16(af[ks], loadB(W, K, n0, ks << 5, lane), c);
    int nl = n0 + (lane & 15);
    float bb = bias[nl];
    int mr = mb + ((lane >> 4) << 3);
#pragma unroll
    for (int r = 0; r < 8; ++r)
      sD[(mr + r) * 512 + nl] = f2us(c[r] + bb);
  }
}

// ================= prep kernels =================
__global__ void conv_bf16_k(const float* __restrict__ s, unsigned short* __restrict__ d, int n){
  for (int i = blockIdx.x * blockDim.x + threadIdx.x; i < n; i += gridDim.x * blockDim.x)
    d[i] = f2us(s[i]);
}
__global__ void tconv_bf16_k(const float* __restrict__ s, int R, int C,
                             unsigned short* __restrict__ d){
  int n = R * C;
  for (int i = blockIdx.x * blockDim.x + threadIdx.x; i < n; i += gridDim.x * blockDim.x){
    int r = i / C, c = i - r * C;
    d[(size_t)c * R + r] = f2us(s[i]);
  }
}
// C[M,N] (bf16) = A[M,512](bf16) @ Bw[N,512](bf16)^T ; one 16x16 tile per wave
__global__ __launch_bounds__(256) void prep_gemm_k(const unsigned short* __restrict__ A,
                                                   const unsigned short* __restrict__ Bw,
                                                   unsigned short* __restrict__ Cc,
                                                   int M, int N){
  int lane = threadIdx.x & 31;
  int gwave = (blockIdx.x * blockDim.x + threadIdx.x) >> 5;
  int nt = N >> 4;
  int tiles = (M >> 4) * nt;
  if (gwave >= tiles) return;           // wave-uniform
  int tm = (gwave / nt) << 4, tn = (gwave % nt) << 4;
  v8f c = {};
#pragma unroll
  for (int ks = 0; ks < 16; ++ks)
    c = wmma_bf16(loadA(A, 512, tm, ks << 5, lane), loadB(Bw, 512, tn, ks << 5, lane), c);
  int nl = tn + (lane & 15);
  int mr = tm + ((lane >> 4) << 3);
#pragma unroll
  for (int r = 0; r < 8; ++r)
    Cc[(size_t)(mr + r) * N + nl] = f2us(c[r]);
}
// folded biases (tiny matvecs)
__global__ void prep_bias_k(const float* __restrict__ Wi, const float* __restrict__ bi,
                            const float* __restrict__ bn, const float* __restrict__ be,
                            const float* __restrict__ W1f, const float* __restrict__ bo,
                            const float* __restrict__ b1f,
                            float* __restrict__ bKn, float* __restrict__ bQe,
                            float* __restrict__ bKe, float* __restrict__ bVn,
                            float* __restrict__ dbV, float* __restrict__ cb1){
  int t = blockIdx.x * blockDim.x + threadIdx.x;
  if (t < 512){
    int i = t; float a = 0.f;
    for (int d = 0; d < 512; ++d) a += Wi[(size_t)(512 + i) * 512 + d] * bn[512 + d];
    bKn[i] = a + bi[512 + i];
  } else if (t < 1024){
    int i = t - 512; float a = 0.f;
    for (int d = 0; d < 512; ++d) a += Wi[(size_t)i * 512 + d] * be[d];
    bQe[i] = a + bi[i];
  } else if (t < 1536){
    int i = t - 1024; float a = 0.f;
    for (int d = 0; d < 512; ++d) a += Wi[(size_t)(512 + i) * 512 + d] * be[512 + d];
    bKe[i] = a + bi[512 + i];
  } else if (t < 2048){
    int i = t - 1536; float a = 0.f;
    for (int d = 0; d < 512; ++d) a += Wi[(size_t)(1024 + i) * 512 + d] * bn[1024 + d];
    bVn[i] = a + bi[1024 + i];
  } else if (t < 2560){
    int i = t - 2048; float a = 0.f;
    for (int d = 0; d < 512; ++d) a += Wi[(size_t)(1024 + i) * 512 + d] * (be[1024 + d] - bn[1024 + d]);
    dbV[i] = a;
  } else if (t < 2816){
    int i = t - 2560; float a = 0.f;
    for (int d = 0; d < 512; ++d) a += W1f[(size_t)i * 512 + d] * bo[d];
    cb1[i] = a + b1f[i];
  }
}

// ================= main kernel 1: Q/K projections + softmax weights =================
__global__ __launch_bounds__(256) void k1_scores(
    const float* __restrict__ U, const float* __restrict__ V, const float* __restrict__ Eg,
    const unsigned short* __restrict__ CQe, const unsigned short* __restrict__ CKn,
    const unsigned short* __restrict__ CKe,
    const float* __restrict__ bQe, const float* __restrict__ bKn, const float* __restrict__ bKe,
    float* __restrict__ attnWS){
  __shared__ alignas(16) unsigned short sU[BM * 256];
  __shared__ alignas(16) unsigned short sV[BM * 256];
  __shared__ alignas(16) unsigned short sE[BM * 128];
  __shared__ alignas(16) unsigned short sQ[BM * 512];
  __shared__ alignas(16) unsigned short sK[BM * 512];
  __shared__ float sSc[3][BM][2];
  int tid = threadIdx.x, lane = tid & 31, wave = tid >> 5;
  int row0 = blockIdx.x * BM;
  load_tile(U, row0, 256, sU, tid);
  load_tile(V, row0, 256, sV, tid);
  load_tile(Eg, row0, 128, sE, tid);
  __syncthreads();
  gemm_tile_to_lds<128>(sE, CQe, bQe, sQ, wave, lane);   // Qe = CQe@e + bQe
  __syncthreads();
#pragma unroll 1
  for (int s = 0; s < 3; ++s){
    if (s == 0)      gemm_tile_to_lds<256>(sU, CKn, bKn, sK, wave, lane);  // K_u
    else if (s == 1) gemm_tile_to_lds<256>(sV, CKn, bKn, sK, wave, lane);  // K_v
    else             gemm_tile_to_lds<128>(sE, CKe, bKe, sK, wave, lane);  // K_e
    __syncthreads();
    if (tid < 128){                                       // per-(row,head) dot Qe.K_s
      int row = tid >> 1, h = tid & 1;
      const uint4* q4 = (const uint4*)(sQ + row * 512 + h * 256);
      const uint4* k4 = (const uint4*)(sK + row * 512 + h * 256);
      float acc = 0.f;
#pragma unroll 4
      for (int i = 0; i < 32; ++i){
        uint4 a = q4[i], b = k4[i];
        acc += us2f(a.x) * us2f(b.x) + us2f(a.x >> 16) * us2f(b.x >> 16);
        acc += us2f(a.y) * us2f(b.y) + us2f(a.y >> 16) * us2f(b.y >> 16);
        acc += us2f(a.z) * us2f(b.z) + us2f(a.z >> 16) * us2f(b.z >> 16);
        acc += us2f(a.w) * us2f(b.w) + us2f(a.w >> 16) * us2f(b.w >> 16);
      }
      sSc[s][row][h] = acc;
    }
    __syncthreads();
  }
  if (tid < 128){                                         // softmax over 3 tokens
    int row = tid >> 1, h = tid & 1;
    float x0 = sSc[0][row][h] * 0.0625f;                  // 1/sqrt(256)
    float x1 = sSc[1][row][h] * 0.0625f;
    float x2 = sSc[2][row][h] * 0.0625f;
    float m = fmaxf(x0, fmaxf(x1, x2));
    float e0 = __expf(x0 - m), e1 = __expf(x1 - m), e2 = __expf(x2 - m);
    float inv = 1.f / (e0 + e1 + e2);
    float* o = attnWS + (size_t)(row0 + row) * 8 + h * 3;
    o[0] = e0 * inv; o[1] = e1 * inv; o[2] = e2 * inv;
  }
}

// ================= main kernel 2: blended V path + folded MLP =================
__global__ __launch_bounds__(256) void k2_output(
    const float* __restrict__ U, const float* __restrict__ V, const float* __restrict__ Eg,
    const float* __restrict__ attnWS,
    const unsigned short* __restrict__ CVn, const unsigned short* __restrict__ CVe,
    const unsigned short* __restrict__ CW1, const unsigned short* __restrict__ W2b,
    const float* __restrict__ bVn, const float* __restrict__ dbV,
    const float* __restrict__ cb1, const float* __restrict__ b2f,
    float* __restrict__ outp){
  __shared__ alignas(16) unsigned short smem[90112];      // 176 KB, aliased regions
  __shared__ float sAttn[BM * 8];
  unsigned short* sU = smem;
  unsigned short* sV = smem + 16384;
  unsigned short* sE = smem + 32768;
  unsigned short* z0 = smem + 40960;
  unsigned short* z1 = smem + 57344;
  unsigned short* e0 = smem + 73728;
  unsigned short* e1 = smem + 81920;
  unsigned short* sO = smem;            // reuse sU+sV after blend
  unsigned short* sH1 = smem + 40960;   // reuse z0 after o-GEMM
  int tid = threadIdx.x, lane = tid & 31, wave = tid >> 5;
  int row0 = blockIdx.x * BM;
  load_tile(U, row0, 256, sU, tid);
  load_tile(V, row0, 256, sV, tid);
  load_tile(Eg, row0, 128, sE, tid);
  for (int i = tid; i < BM * 8; i += 256) sAttn[i] = attnWS[(size_t)row0 * 8 + i];
  __syncthreads();
  // per-head blended inputs: z_h = a_hu*u + a_hv*v ; e_h = a_he*e
  for (int i = tid; i < BM * 128; i += 256){
    int row = i >> 7;
    unsigned int uu = ((const unsigned int*)sU)[i];
    unsigned int vv = ((const unsigned int*)sV)[i];
    float u0 = us2f(uu), u1 = us2f(uu >> 16);
    float v0 = us2f(vv), v1 = us2f(vv >> 16);
    float a0u = sAttn[row * 8 + 0], a0v = sAttn[row * 8 + 1];
    float a1u = sAttn[row * 8 + 3], a1v = sAttn[row * 8 + 4];
    ((unsigned int*)z0)[i] = pk2(a0u * u0 + a0v * v0, a0u * u1 + a0v * v1);
    ((unsigned int*)z1)[i] = pk2(a1u * u0 + a1v * v0, a1u * u1 + a1v * v1);
  }
  for (int i = tid; i < BM * 64; i += 256){
    int row = i >> 6;
    unsigned int ee = ((const unsigned int*)sE)[i];
    float f0 = us2f(ee), f1 = us2f(ee >> 16);
    float a0e = sAttn[row * 8 + 2], a1e = sAttn[row * 8 + 5];
    ((unsigned int*)e0)[i] = pk2(a0e * f0, a0e * f1);
    ((unsigned int*)e1)[i] = pk2(a1e * f0, a1e * f1);
  }
  __syncthreads();
  // o_h = CVn_h @ z_h + CVe_h @ e_h + bVn + a_e*(bVe-bVn)
  {
    int h = wave >> 2, mb = (wave & 3) << 4;
    const unsigned short* zh = h ? z1 : z0;
    const unsigned short* eh = h ? e1 : e0;
    const unsigned short* Wz = CVn + (size_t)h * 256 * 256;
    const unsigned short* Wee = CVe + (size_t)h * 256 * 128;
    v16bf az[8], ax[4];
#pragma unroll
    for (int ks = 0; ks < 8; ++ks) az[ks] = loadA(zh, 256, mb, ks << 5, lane);
#pragma unroll
    for (int ks = 0; ks < 4; ++ks) ax[ks] = loadA(eh, 128, mb, ks << 5, lane);
#pragma unroll 4
    for (int nb = 0; nb < 16; ++nb){
      int n0 = nb << 4;
      v8f c = {};
#pragma unroll
      for (int ks = 0; ks < 8; ++ks) c = wmma_bf16(az[ks], loadB(Wz, 256, n0, ks << 5, lane), c);
#pragma unroll
      for (int ks = 0; ks < 4; ++ks) c = wmma_bf16(ax[ks], loadB(Wee, 128, n0, ks << 5, lane), c);
      int nl = n0 + (lane & 15);
      int ng = (h << 8) + nl;
      float bv = bVn[ng], dv = dbV[ng];
      int mr = mb + ((lane >> 4) << 3);
#pragma unroll
      for (int r = 0; r < 8; ++r){
        int row = mr + r;
        float a_e = sAttn[row * 8 + h * 3 + 2];
        sO[row * 512 + ng] = f2us(c[r] + bv + a_e * dv);
      }
    }
  }
  __syncthreads();
  // h1 = silu(sO @ CW1^T + cb1)   (CW1 = W1@Wo)
  {
    int mb = (wave & 3) << 4, nh = wave >> 2;
    v16bf af[16];
#pragma unroll
    for (int ks = 0; ks < 16; ++ks) af[ks] = loadA(sO, 512, mb, ks << 5, lane);
#pragma unroll 2
    for (int nb = nh << 3; nb < (nh << 3) + 8; ++nb){
      int n0 = nb << 4;
      v8f c = {};
#pragma unroll
      for (int ks = 0; ks < 16; ++ks) c = wmma_bf16(af[ks], loadB(CW1, 512, n0, ks << 5, lane), c);
      int nl = n0 + (lane & 15);
      float bb = cb1[nl];
      int mr = mb + ((lane >> 4) << 3);
#pragma unroll
      for (int r = 0; r < 8; ++r){
        float x = c[r] + bb;
        sH1[(mr + r) * 256 + nl] = f2us(x * (1.f / (1.f + __expf(-x))));
      }
    }
  }
  __syncthreads();
  // out = sH1 @ W2^T + b2  (fp32 stores)
  {
    int mb = (wave & 3) << 4, nq = wave >> 2;
    v16bf af[8];
#pragma unroll
    for (int ks = 0; ks < 8; ++ks) af[ks] = loadA(sH1, 256, mb, ks << 5, lane);
#pragma unroll
    for (int nb = nq << 2; nb < (nq << 2) + 4; ++nb){
      int n0 = nb << 4;
      v8f c = {};
#pragma unroll
      for (int ks = 0; ks < 8; ++ks) c = wmma_bf16(af[ks], loadB(W2b, 256, n0, ks << 5, lane), c);
      int nl = n0 + (lane & 15);
      float bb = b2f[nl];
      int mr = mb + ((lane >> 4) << 3);
#pragma unroll
      for (int r = 0; r < 8; ++r)
        outp[(size_t)(row0 + mr + r) * 128 + nl] = c[r] + bb;
    }
  }
}

// ================= launch =================
extern "C" void kernel_launch(void* const* d_in, const int* in_sizes, int n_in,
                              void* d_out, int out_size, void* d_ws, size_t ws_size,
                              hipStream_t stream){
  const float* node_us = (const float*)d_in[0];
  const float* node_vs = (const float*)d_in[1];
  const float* edges   = (const float*)d_in[2];
  const float* Wn = (const float*)d_in[3];
  const float* bn = (const float*)d_in[4];
  const float* We = (const float*)d_in[5];
  const float* be = (const float*)d_in[6];
  const float* Wi = (const float*)d_in[7];
  const float* bi = (const float*)d_in[8];
  const float* Wo = (const float*)d_in[9];
  const float* bo = (const float*)d_in[10];
  const float* W1 = (const float*)d_in[11];
  const float* b1 = (const float*)d_in[12];
  const float* W2 = (const float*)d_in[13];
  const float* b2 = (const float*)d_in[14];
  float* outp = (float*)d_out;
  (void)in_sizes; (void)n_in; (void)out_size; (void)ws_size;

  char* base = (char*)d_ws;
  size_t off = 0;
  auto carve = [&](size_t bytes) -> char* {
    char* p = base + off;
    off += (bytes + 255) & ~(size_t)255;
    return p;
  };
  unsigned short* Wi_b = (unsigned short*)carve(1536 * 512 * 2);
  unsigned short* W1_b = (unsigned short*)carve(256 * 512 * 2);
  unsigned short* W2b  = (unsigned short*)carve(128 * 256 * 2);
  unsigned short* WnkT = (unsigned short*)carve(256 * 512 * 2);
  unsigned short* WnvT = (unsigned short*)carve(256 * 512 * 2);
  unsigned short* WeqT = (unsigned short*)carve(128 * 512 * 2);
  unsigned short* WekT = (unsigned short*)carve(128 * 512 * 2);
  unsigned short* WevT = (unsigned short*)carve(128 * 512 * 2);
  unsigned short* WoT  = (unsigned short*)carve(512 * 512 * 2);
  unsigned short* CKn  = (unsigned short*)carve(512 * 256 * 2);
  unsigned short* CVn  = (unsigned short*)carve(512 * 256 * 2);
  unsigned short* CQe  = (unsigned short*)carve(512 * 128 * 2);
  unsigned short* CKe  = (unsigned short*)carve(512 * 128 * 2);
  unsigned short* CVe  = (unsigned short*)carve(512 * 128 * 2);
  unsigned short* CW1  = (unsigned short*)carve(256 * 512 * 2);
  float* bKn = (float*)carve(512 * 4);
  float* bQe = (float*)carve(512 * 4);
  float* bKe = (float*)carve(512 * 4);
  float* bVn = (float*)carve(512 * 4);
  float* dbV = (float*)carve(512 * 4);
  float* cb1 = (float*)carve(256 * 4);
  float* attn = (float*)carve((size_t)B_ROWS * 8 * 4);

  // weight converts / transposes (fp32 -> bf16)
  conv_bf16_k<<<256, 256, 0, stream>>>(Wi, Wi_b, 1536 * 512);
  conv_bf16_k<<<64, 256, 0, stream>>>(W1, W1_b, 256 * 512);
  conv_bf16_k<<<32, 256, 0, stream>>>(W2, W2b, 128 * 256);
  tconv_bf16_k<<<128, 256, 0, stream>>>(Wn + 512 * 256, 512, 256, WnkT);
  tconv_bf16_k<<<128, 256, 0, stream>>>(Wn + 1024 * 256, 512, 256, WnvT);
  tconv_bf16_k<<<64, 256, 0, stream>>>(We, 512, 128, WeqT);
  tconv_bf16_k<<<64, 256, 0, stream>>>(We + 512 * 128, 512, 128, WekT);
  tconv_bf16_k<<<64, 256, 0, stream>>>(We + 1024 * 128, 512, 128, WevT);
  tconv_bf16_k<<<256, 256, 0, stream>>>(Wo, 512, 512, WoT);
  // composite weights via WMMA
  prep_gemm_k<<<64, 256, 0, stream>>>(Wi_b + 512 * 512, WnkT, CKn, 512, 256);   // Wk@Wn_k
  prep_gemm_k<<<64, 256, 0, stream>>>(Wi_b + 1024 * 512, WnvT, CVn, 512, 256);  // Wv@Wn_v
  prep_gemm_k<<<32, 256, 0, stream>>>(Wi_b, WeqT, CQe, 512, 128);               // Wq@We_q
  prep_gemm_k<<<32, 256, 0, stream>>>(Wi_b + 512 * 512, WekT, CKe, 512, 128);   // Wk@We_k
  prep_gemm_k<<<32, 256, 0, stream>>>(Wi_b + 1024 * 512, WevT, CVe, 512, 128);  // Wv@We_v
  prep_gemm_k<<<64, 256, 0, stream>>>(W1_b, WoT, CW1, 256, 512);                // W1@Wo
  prep_bias_k<<<11, 256, 0, stream>>>(Wi, bi, bn, be, W1, bo, b1,
                                      bKn, bQe, bKe, bVn, dbV, cb1);
  // main fused pipeline
  k1_scores<<<B_ROWS / BM, 256, 0, stream>>>(node_us, node_vs, edges,
                                             CQe, CKn, CKe, bQe, bKn, bKe, attn);
  k2_output<<<B_ROWS / BM, 256, 0, stream>>>(node_us, node_vs, edges, attn,
                                             CVn, CVe, CW1, W2b,
                                             bVn, dbV, cb1, b2, outp);
}
